// GlobalAttention_90434831385120
// MI455X (gfx1250) — matrix operand order repaired and checked by reading, not compiled, not run on hardware
//
#include <hip/hip_runtime.h>
#include <hip/hip_bf16.h>

typedef __attribute__((ext_vector_type(16))) __bf16 v16bf;
typedef __attribute__((ext_vector_type(8)))  float  v8f;

// ---------------------------------------------------------------- helpers ---

__device__ __forceinline__ v8f wmma_bf16(v16bf a, v16bf b, v8f c) {
  // D = A(16x32 bf16) * B(32x16 bf16) + C(16x16 f32)
  return __builtin_amdgcn_wmma_f32_16x16x32_bf16(
      false, a, false, b, (short)0, c, false, false);
}

// wave-internal LDS producer->consumer ordering (single PC per wave + in-order
// LDS pipe; the wait drains DScnt, the clobber fences the compiler)
__device__ __forceinline__ void wave_lds_fence() {
  asm volatile("s_wait_dscnt 0" ::: "memory");
}

// A-matrix fragment (16-bit, 16xK): lane holds row M=lane&15.
// VGPR j<4 : K = 2j   + 8*hi ; VGPR j>=4 : K = 16+2(j-4) + 8*hi   (hi=lane>>4)
// kvalid in [1,32]: K positions >= kvalid are zero-padded; memory K = k0+kb.
__device__ __forceinline__ v16bf load_a_frag(const __bf16* T, int stride,
                                             int lane, int k0, int kvalid) {
  v16bf a;
  const int M  = lane & 15;
  const int hi = (lane >> 4) & 1;
#pragma unroll
  for (int j = 0; j < 8; ++j) {
    int kb = ((j < 4) ? (2 * j) : (16 + 2 * (j - 4))) + 8 * hi;
    int k  = k0 + kb;
    a[2 * j]     = (kb     < kvalid) ? T[M * stride + k]     : (__bf16)0.0f;
    a[2 * j + 1] = (kb + 1 < kvalid) ? T[M * stride + k + 1] : (__bf16)0.0f;
  }
  return a;
}

// B-matrix fragment (16-bit, Kx16) gathered from a [N][K] row-major table:
// lane holds column N = n0 + (lane&15); VGPR j<4 : K = 2j + 8*hi ; VGPR j>=4 : K = 16+2(j-4) + 8*hi.
__device__ __forceinline__ v16bf load_b_frag(const __bf16* T, int stride,
                                             int lane, int k0, int n0,
                                             int kvalid) {
  v16bf b;
  const int n  = n0 + (lane & 15);
  const int hi = (lane >> 4) & 1;
#pragma unroll
  for (int j = 0; j < 8; ++j) {
    int kb = ((j < 4) ? (2 * j) : (16 + 2 * (j - 4))) + 8 * hi;
    int k  = k0 + kb;
    b[2 * j]     = (kb     < kvalid) ? T[n * stride + k]     : (__bf16)0.0f;
    b[2 * j + 1] = (kb + 1 < kvalid) ? T[n * stride + k + 1] : (__bf16)0.0f;
  }
  return b;
}

// reductions across the 16 lanes holding one C/D row (masks stay in half-wave)
__device__ __forceinline__ float rowmax16(float v) {
#pragma unroll
  for (int d = 1; d < 16; d <<= 1) v = fmaxf(v, __shfl_xor(v, d, 32));
  return v;
}
__device__ __forceinline__ float rowsum16(float v) {
#pragma unroll
  for (int d = 1; d < 16; d <<= 1) v += __shfl_xor(v, d, 32);
  return v;
}

// stage a [rows x 64] fp32 weight into LDS as bf16 with row stride 66
__device__ __forceinline__ void stage_w(const float* __restrict__ w,
                                        __bf16* dst, int rows) {
  for (int i = threadIdx.x; i < rows * 64; i += blockDim.x) {
    int r = i >> 6, c = i & 63;
    dst[r * 66 + c] = (__bf16)w[i];
  }
}

// ------------------------------------------------ stage 1: 4x4 window attn --

__global__ __launch_bounds__(256) void attn_win4_kernel(
    const float* __restrict__ x, const float* __restrict__ wq,
    const float* __restrict__ bq, const float* __restrict__ wk,
    const float* __restrict__ bk, const float* __restrict__ wv,
    const float* __restrict__ bv, const float* __restrict__ gamma,
    float* __restrict__ out1) {
  __shared__ __bf16 sWq[16 * 66];
  __shared__ __bf16 sWk[16 * 66];
  __shared__ __bf16 sWv[64 * 66];
  __shared__ __bf16 sX[8][16 * 66];   // [pixel][ch]
  __shared__ __bf16 sQ[8][16 * 18];   // [pixel][cq]
  __shared__ __bf16 sK[8][16 * 18];   // [pixel][cq]
  __shared__ __bf16 sVt[8][64 * 18];  // [ch][pixel]
  __shared__ __bf16 sP[8][16 * 18];   // [row][col]

  stage_w(wq, sWq, 16);
  stage_w(wk, sWk, 16);
  stage_w(wv, sWv, 64);
  __syncthreads();

  const int lane  = threadIdx.x & 31;
  const int wavei = threadIdx.x >> 5;
  const int col   = lane & 15;
  const int hi    = (lane >> 4) & 1;

  const int win  = blockIdx.x * 8 + wavei;  // 2304 blocks * 8 = 18432 windows
  const int bimg = win / 2304;
  const int rem  = win - bimg * 2304;
  const int wh   = rem / 48;
  const int ww   = rem - wh * 48;

  __bf16* Xw  = sX[wavei];
  __bf16* Qw  = sQ[wavei];
  __bf16* Kw  = sK[wavei];
  __bf16* Vtw = sVt[wavei];
  __bf16* Pw  = sP[wavei];

  // ---- load 16-pixel x 64-ch window, fp32 -> bf16 into LDS
#pragma unroll
  for (int cc = 0; cc < 2; ++cc) {
    int ch = lane + 32 * cc;
    const float* src =
        x + ((size_t)((bimg * 64 + ch) * 192 + wh * 4)) * 192 + ww * 4;
#pragma unroll
    for (int i = 0; i < 4; ++i)
#pragma unroll
      for (int j = 0; j < 4; ++j)
        Xw[(i * 4 + j) * 66 + ch] = (__bf16)src[i * 192 + j];
  }
  wave_lds_fence();

  v16bf ax0 = load_a_frag(Xw, 66, lane, 0, 32);
  v16bf ax1 = load_a_frag(Xw, 66, lane, 32, 32);

  // ---- Q = X*Wq^T + bq ; K = X*Wk^T + bk     (16x16, K over 64 channels)
  {
    v8f q = {};
    q = wmma_bf16(ax0, load_b_frag(sWq, 66, lane, 0, 0, 32), q);
    q = wmma_bf16(ax1, load_b_frag(sWq, 66, lane, 32, 0, 32), q);
    float bqv = bq[col];
    v8f k = {};
    k = wmma_bf16(ax0, load_b_frag(sWk, 66, lane, 0, 0, 32), k);
    k = wmma_bf16(ax1, load_b_frag(sWk, 66, lane, 32, 0, 32), k);
    float bkv = bk[col];
#pragma unroll
    for (int g = 0; g < 8; ++g) {
      int M = g + 8 * hi;
      Qw[M * 18 + col] = (__bf16)(q[g] + bqv);
      Kw[M * 18 + col] = (__bf16)(k[g] + bkv);
    }
  }
  // ---- V = X*Wv^T + bv  (16x64), stored transposed [ch][pixel]
#pragma unroll
  for (int nc = 0; nc < 4; ++nc) {
    v8f v = {};
    v = wmma_bf16(ax0, load_b_frag(sWv, 66, lane, 0, nc * 16, 32), v);
    v = wmma_bf16(ax1, load_b_frag(sWv, 66, lane, 32, nc * 16, 32), v);
    int ch    = nc * 16 + col;
    float bvv = bv[ch];
#pragma unroll
    for (int g = 0; g < 8; ++g) Vtw[ch * 18 + (g + 8 * hi)] = (__bf16)(v[g] + bvv);
  }
  wave_lds_fence();

  // ---- S = Q * K^T (16x16, K=Cq=16 zero-padded to 32), softmax rows
  {
    v16bf aq = load_a_frag(Qw, 18, lane, 0, 16);
    v16bf bk_ = load_b_frag(Kw, 18, lane, 0, 0, 16);
    v8f z = {};
    v8f s = wmma_bf16(aq, bk_, z);
#pragma unroll
    for (int g = 0; g < 8; ++g) {
      float mx = rowmax16(s[g]);
      float p  = __expf(s[g] - mx);
      float sm = rowsum16(p);
      Pw[(g + 8 * hi) * 18 + col] = (__bf16)(p / sm);
    }
  }
  wave_lds_fence();

  // ---- Out = P * V ; out1 = gamma*Out + x
  {
    const float g1 = gamma[0];
    v16bf ap = load_a_frag(Pw, 18, lane, 0, 16);
#pragma unroll
    for (int nc = 0; nc < 4; ++nc) {
      v8f z = {};
      v8f o = wmma_bf16(ap, load_b_frag(Vtw, 18, lane, 0, nc * 16, 16), z);
      int ch = nc * 16 + col;
#pragma unroll
      for (int g = 0; g < 8; ++g) {
        int M = g + 8 * hi;
        size_t idx = ((size_t)((bimg * 64 + ch) * 192 + wh * 4 + (M >> 2))) * 192 +
                     (ww * 4 + (M & 3));
        out1[idx] = g1 * o[g] + x[idx];
      }
    }
  }
}

// ------------------------------------- stage 2: chess (win=8) flash attn ----

__device__ __forceinline__ void load_xtile_chess(const float* __restrict__ x1,
                                                 __bf16* scr, int lane,
                                                 int bimg, int aoff, int cf,
                                                 int t) {
#pragma unroll
  for (int cc = 0; cc < 2; ++cc) {
    int ch = lane + 32 * cc;
    const float* base = x1 + (size_t)(bimg * 64 + ch) * 192 * 192;
#pragma unroll
    for (int m = 0; m < 16; ++m) {
      int n = t * 16 + m;
      int p = n / 24, qq = n - p * 24;
      int h = p * 8 + aoff, w = qq * 8 + cf;
      scr[m * 66 + ch] = (__bf16)base[h * 192 + w];
    }
  }
}

__global__ __launch_bounds__(256) void attn_chess8_kernel(
    const float* __restrict__ x1, const float* __restrict__ wq,
    const float* __restrict__ bq, const float* __restrict__ wk,
    const float* __restrict__ bk, const float* __restrict__ wv,
    const float* __restrict__ bv, const float* __restrict__ gamma,
    float* __restrict__ out) {
  __shared__ __bf16 sWq[16 * 66];
  __shared__ __bf16 sWk[16 * 66];
  __shared__ __bf16 sWv[64 * 66];
  __shared__ __bf16 sKm[576 * 18];     // [m][cq]
  __shared__ __bf16 sVt[64 * 578];     // [ch][m]
  __shared__ __bf16 sScr[8][16 * 66];  // per-wave X / Q / P scratch

  const int grp  = blockIdx.x;  // 512 groups
  const int bimg = grp >> 6;
  const int aoff = (grp >> 3) & 7;
  const int cf   = grp & 7;

  stage_w(wq, sWq, 16);
  stage_w(wk, sWk, 16);
  stage_w(wv, sWv, 64);
  __syncthreads();

  const int lane  = threadIdx.x & 31;
  const int wavei = threadIdx.x >> 5;
  const int col   = lane & 15;
  const int hi    = (lane >> 4) & 1;
  __bf16* scr = sScr[wavei];

  // ---- phase 1: K,V for all 576 pixels (8 waves, 16-pixel tiles)
  for (int t = wavei; t < 36; t += 8) {
    load_xtile_chess(x1, scr, lane, bimg, aoff, cf, t);
    wave_lds_fence();
    v16bf ax0 = load_a_frag(scr, 66, lane, 0, 32);
    v16bf ax1 = load_a_frag(scr, 66, lane, 32, 32);

    v8f kacc = {};
    kacc = wmma_bf16(ax0, load_b_frag(sWk, 66, lane, 0, 0, 32), kacc);
    kacc = wmma_bf16(ax1, load_b_frag(sWk, 66, lane, 32, 0, 32), kacc);
    float bkv = bk[col];
#pragma unroll
    for (int g = 0; g < 8; ++g)
      sKm[(t * 16 + g + 8 * hi) * 18 + col] = (__bf16)(kacc[g] + bkv);

#pragma unroll
    for (int nc = 0; nc < 4; ++nc) {
      v8f vacc = {};
      vacc = wmma_bf16(ax0, load_b_frag(sWv, 66, lane, 0, nc * 16, 32), vacc);
      vacc = wmma_bf16(ax1, load_b_frag(sWv, 66, lane, 32, nc * 16, 32), vacc);
      int ch    = nc * 16 + col;
      float bvv = bv[ch];
#pragma unroll
      for (int g = 0; g < 8; ++g)
        sVt[ch * 578 + t * 16 + g + 8 * hi] = (__bf16)(vacc[g] + bvv);
    }
    wave_lds_fence();
  }
  __syncthreads();

  // ---- phase 2: flash attention, 16-row query tiles, 32-wide key chunks
  const float g2 = gamma[0];
  for (int t = wavei; t < 36; t += 8) {
    load_xtile_chess(x1, scr, lane, bimg, aoff, cf, t);
    wave_lds_fence();
    v16bf ax0 = load_a_frag(scr, 66, lane, 0, 32);
    v16bf ax1 = load_a_frag(scr, 66, lane, 32, 32);
    v8f qacc = {};
    qacc = wmma_bf16(ax0, load_b_frag(sWq, 66, lane, 0, 0, 32), qacc);
    qacc = wmma_bf16(ax1, load_b_frag(sWq, 66, lane, 32, 0, 32), qacc);
    float bqv = bq[col];
#pragma unroll
    for (int g = 0; g < 8; ++g)
      scr[(g + 8 * hi) * 66 + col] = (__bf16)(qacc[g] + bqv);  // Q -> cols 0..15
    wave_lds_fence();
    v16bf aq = load_a_frag(scr, 66, lane, 0, 16);

    float rm[8], ls[8];
    v8f O0 = {}, O1 = {}, O2 = {}, O3 = {};
#pragma unroll
    for (int g = 0; g < 8; ++g) {
      rm[g] = -__builtin_inff();
      ls[g] = 0.0f;
    }

    for (int mc = 0; mc < 18; ++mc) {
      v8f z = {};
      v8f s0 = wmma_bf16(aq, load_b_frag(sKm, 18, lane, 0, mc * 32, 16), z);
      v8f s1 = wmma_bf16(aq, load_b_frag(sKm, 18, lane, 0, mc * 32 + 16, 16), z);
#pragma unroll
      for (int g = 0; g < 8; ++g) {
        float mx = rowmax16(fmaxf(s0[g], s1[g]));
        float nm = fmaxf(rm[g], mx);
        float al = __expf(rm[g] - nm);
        rm[g]    = nm;
        float p0 = __expf(s0[g] - nm);
        float p1 = __expf(s1[g] - nm);
        ls[g]    = ls[g] * al + rowsum16(p0 + p1);
        O0[g] *= al; O1[g] *= al; O2[g] *= al; O3[g] *= al;
        int M = g + 8 * hi;
        scr[M * 66 + 32 + col] = (__bf16)p0;  // P -> cols 32..63
        scr[M * 66 + 48 + col] = (__bf16)p1;
      }
      wave_lds_fence();
      v16bf ap = load_a_frag(scr + 32, 66, lane, 0, 32);
      O0 = wmma_bf16(ap, load_b_frag(sVt, 578, lane, mc * 32, 0, 32), O0);
      O1 = wmma_bf16(ap, load_b_frag(sVt, 578, lane, mc * 32, 16, 32), O1);
      O2 = wmma_bf16(ap, load_b_frag(sVt, 578, lane, mc * 32, 32, 32), O2);
      O3 = wmma_bf16(ap, load_b_frag(sVt, 578, lane, mc * 32, 48, 32), O3);
      wave_lds_fence();
    }

    v8f Oa[4] = {O0, O1, O2, O3};
#pragma unroll
    for (int nc = 0; nc < 4; ++nc) {
      int ch = nc * 16 + col;
#pragma unroll
      for (int g = 0; g < 8; ++g) {
        int M = g + 8 * hi;
        int n = t * 16 + M;
        int p = n / 24, qq = n - p * 24;
        int h = p * 8 + aoff, w = qq * 8 + cf;
        size_t idx = ((size_t)((bimg * 64 + ch) * 192 + h)) * 192 + w;
        out[idx] = g2 * (Oa[nc][g] / ls[g]) + x1[idx];
      }
    }
  }
}

// ---------------------------------------------------------------- launcher --

extern "C" void kernel_launch(void* const* d_in, const int* in_sizes, int n_in,
                              void* d_out, int out_size, void* d_ws,
                              size_t ws_size, hipStream_t stream) {
  (void)in_sizes; (void)n_in; (void)out_size; (void)ws_size;
  const float* x   = (const float*)d_in[0];
  const float* wq1 = (const float*)d_in[1];
  const float* bq1 = (const float*)d_in[2];
  const float* wk1 = (const float*)d_in[3];
  const float* bk1 = (const float*)d_in[4];
  const float* wv1 = (const float*)d_in[5];
  const float* bv1 = (const float*)d_in[6];
  const float* wq2 = (const float*)d_in[7];
  const float* bq2 = (const float*)d_in[8];
  const float* wk2 = (const float*)d_in[9];
  const float* bk2 = (const float*)d_in[10];
  const float* wv2 = (const float*)d_in[11];
  const float* bv2 = (const float*)d_in[12];
  const float* g1  = (const float*)d_in[13];
  const float* g2  = (const float*)d_in[14];

  float* out1 = (float*)d_ws;  // [8,64,192,192] fp32 intermediate

  attn_win4_kernel<<<2304, 256, 0, stream>>>(x, wq1, bq1, wk1, bk1, wv1, bv1,
                                             g1, out1);
  attn_chess8_kernel<<<512, 256, 0, stream>>>(out1, wq2, bq2, wk2, bk2, wv2,
                                              bv2, g2, (float*)d_out);
}
